// Network_79053168050265
// MI455X (gfx1250) — matrix operand order repaired and checked by reading, not compiled, not run on hardware
//
#include <hip/hip_runtime.h>

typedef __attribute__((ext_vector_type(16))) _Float16 v16h;
typedef __attribute__((ext_vector_type(8)))  _Float16 v8h;
typedef __attribute__((ext_vector_type(8)))  float    v8f;

// global (addrspace 1) views of the packed weights: loads lower to
// global_load_b128 (LOADcnt only) instead of flat_load (LOADcnt+DScnt)
typedef __attribute__((address_space(1))) const _Float16 gf16_t;
typedef __attribute__((address_space(1))) const v16h    gv16h_t;

#define BATCH  256
#define TSTEPS 512
#define FB     128
#define HL1    256
#define HL2    256
#define HL3    128
#define NCLS   10

// ---- packed f16 weight layout in workspace (units: halfs) ----
// Each WMMA tile is 16(N) x 32(K) = 512 halfs, stored so that lane l's
// B-fragment (v16h) is contiguous at tile_base + l*16:
//   n = ntile*16 + (l & 15),  k = ktile*32 + ((l & 16) ? 8 : 0) + (i & 7) + ((i & 8) ? 16 : 0)   (i = 0..15)
#define W1_OFF 0
#define W1_NKT 12                         // K = 384
#define W2_OFF (W1_OFF + 32 * 12 * 512)   // 196608, K = 512, 32 n-tiles
#define W2_NKT 16
#define W3_OFF (W2_OFF + 32 * 16 * 512)   // 458752, K = 256, 8 n-tiles
#define W3_NKT 8
#define W4_OFF (W3_OFF + 8 * 8 * 512)     // 491520, K = 128, 1 n-tile (N padded 10->16)
#define W4_NKT 4
#define WTOTAL (W4_OFF + 1 * 4 * 512)     // 493568 halfs = ~964 KB

// 4-bit fixed point quantizer: clip(round(w*8)/8, -1, 1), round-half-even like jnp.round
__device__ __forceinline__ _Float16 quant_q(float w) {
    float q = rintf(w * 8.0f) * 0.125f;
    q = fminf(fmaxf(q, -1.0f), 1.0f);
    return (_Float16)q;
}

__global__ void pack_weights(const float* __restrict__ W1f, const float* __restrict__ W2f,
                             const float* __restrict__ W3f, const float* __restrict__ W4f,
                             _Float16* __restrict__ wp)
{
    for (int idx = blockIdx.x * blockDim.x + threadIdx.x; idx < WTOTAL;
         idx += gridDim.x * blockDim.x) {
        const float* W; int nkt, K, Nvalid, rel;
        if (idx < W2_OFF)      { W = W1f; nkt = W1_NKT; K = 384; Nvalid = 512; rel = idx - W1_OFF; }
        else if (idx < W3_OFF) { W = W2f; nkt = W2_NKT; K = 512; Nvalid = 512; rel = idx - W2_OFF; }
        else if (idx < W4_OFF) { W = W3f; nkt = W3_NKT; K = 256; Nvalid = 128; rel = idx - W3_OFF; }
        else                   { W = W4f; nkt = W4_NKT; K = 128; Nvalid = 10;  rel = idx - W4_OFF; }
        int tile = rel >> 9, e = rel & 511;
        int lane = e >> 4, i = e & 15;
        int nt = tile / nkt, kt = tile % nkt;
        int n = nt * 16 + (lane & 15);
        int k = kt * 32 + ((lane & 16) ? 8 : 0) + (i & 7) + ((i & 8) ? 16 : 0);
        wp[idx] = (n < Nvalid) ? quant_q(W[(size_t)n * K + k]) : (_Float16)0.0f;
    }
}

// A-fragment (16x32 f16, ISA layout): lane<16 -> row m=lane, K {0..7,16..23};
// lane>=16 -> row m=lane-16, K {8..15,24..31}. Two b128 LDS loads per lane.
union AFrag { v16h v; v8h h[2]; };
__device__ __forceinline__ v16h load_afrag(const _Float16* row, int kt, int lane) {
    int k0 = kt * 32 + ((lane & 16) ? 8 : 0);
    AFrag af;
    af.h[0] = *(const v8h*)(row + k0);
    af.h[1] = *(const v8h*)(row + k0 + 16);
    return af.v;
}

__device__ __forceinline__ v8f wmma_f16(v16h a, v16h b, v8f c) {
    return __builtin_amdgcn_wmma_f32_16x16x32_f16(false, a, false, b, (short)0, c, false, false);
}

// softsign: denominator 1+|x| >= 1, so a ~1ulp v_rcp_f32 is plenty — avoids the
// IEEE v_div_scale/v_div_fmas Newton chain (~20 VALU ops) the plain '/' emitted.
__device__ __forceinline__ float softsign(float x) {
    return x * __builtin_amdgcn_rcpf(1.0f + fabsf(x));
}

// Persistent eGRU scan. grid = 16 workgroups (one 16-row batch tile each),
// block = 512 threads = 16 waves. Each wave owns a matched (z, candidate)
// N-tile pair in the gated layers.
__global__ void __launch_bounds__(512)
egru_forward(const float* __restrict__ x,
             const float* __restrict__ h1i, const float* __restrict__ h2i,
             const float* __restrict__ b1,  const float* __restrict__ b2,
             const float* __restrict__ b3,  const float* __restrict__ b4,
             const _Float16* __restrict__ wp,
             float* __restrict__ out)
{
    __shared__ __align__(16) _Float16 act1[16 * 384];  // [x(128) | h1(256)]
    __shared__ __align__(16) _Float16 act2[16 * 512];  // [h1_new(256) | h2(256)]
    __shared__ __align__(16) _Float16 act3[16 * 256];  // h2_new
    __shared__ __align__(16) _Float16 act4[16 * 128];  // h3
    __shared__ float smax[16 * 16];                    // logits staging

    const int tid   = threadIdx.x;
    const int lane  = tid & 31;
    const int wv    = tid >> 5;          // 0..15
    const int bbase = blockIdx.x * 16;
    const int m_a   = lane & 15;         // A-fragment row
    const int ncol  = lane & 15;         // D column within tile
    const int mbase = (lane & 16) ? 8 : 0;
    const int n_t   = wv * 16 + ncol;    // this wave's gate column (L1/L2)

    // hoist per-lane bias scalars out of the scan (guarded for non-owning waves)
    const float biasZ1 = b1[n_t];
    const float biasC1 = b1[HL1 + n_t];
    const float biasZ2 = b2[n_t];
    const float biasC2 = b2[HL2 + n_t];
    const float bias3  = (wv < 8) ? b3[wv * 16 + ncol] : 0.0f;
    const float bias4  = (wv == 0 && ncol < NCLS) ? b4[ncol] : 0.0f;

#pragma unroll 1
    for (int t = 0; t < TSTEPS; ++t) {
        // Launder the (global-addrspace) weight base pointer through an SGPR pair
        // once per step: every weight load's address is formally loop-variant, so
        // LICM cannot hoist 512-iteration-invariant B-fragments into registers
        // (which spilled to scratch in round 1). The explicit address_space(1)
        // type keeps the loads on the global_load path (round 2 regressed to flat).
        gf16_t* wpt = (gf16_t*)wp;
        asm volatile("" : "+s"(wpt));

        // ---- stage x_t into act1[:, 0:128] (f32 -> f16) ----
        {
            int j = tid * 4;                      // 512 threads * 4 = 2048 = 16*128
            int m = j >> 7, k = j & 127;
            const float* xp = x + ((size_t)(bbase + m) * TSTEPS + t) * FB + k;
            const float4 xv = *(const float4*)xp;
            _Float16* dst = act1 + m * 384 + k;
            dst[0] = (_Float16)xv.x; dst[1] = (_Float16)xv.y;
            dst[2] = (_Float16)xv.z; dst[3] = (_Float16)xv.w;
            if (t + 1 < TSTEPS)                   // overlap next step's HBM stream
                __builtin_prefetch(xp + FB, 0, 1);
        }
        // ---- stage h1 into act1[:,128:384] and h2 into act2[:,256:512] ----
        {
            int j = tid * 8;                      // 512 threads * 8 = 4096 = 16*256
            int m = j >> 8, c = j & 255;
            if (t == 0) {
                const float4 a0 = *(const float4*)(h1i + (size_t)(bbase + m) * HL1 + c);
                const float4 a1 = *(const float4*)(h1i + (size_t)(bbase + m) * HL1 + c + 4);
                _Float16* d1 = act1 + m * 384 + 128 + c;
                d1[0]=(_Float16)a0.x; d1[1]=(_Float16)a0.y; d1[2]=(_Float16)a0.z; d1[3]=(_Float16)a0.w;
                d1[4]=(_Float16)a1.x; d1[5]=(_Float16)a1.y; d1[6]=(_Float16)a1.z; d1[7]=(_Float16)a1.w;
                const float4 e0 = *(const float4*)(h2i + (size_t)(bbase + m) * HL2 + c);
                const float4 e1 = *(const float4*)(h2i + (size_t)(bbase + m) * HL2 + c + 4);
                _Float16* d2 = act2 + m * 512 + 256 + c;
                d2[0]=(_Float16)e0.x; d2[1]=(_Float16)e0.y; d2[2]=(_Float16)e0.z; d2[3]=(_Float16)e0.w;
                d2[4]=(_Float16)e1.x; d2[5]=(_Float16)e1.y; d2[6]=(_Float16)e1.z; d2[7]=(_Float16)e1.w;
            } else {
                *(v8h*)(act1 + m * 384 + 128 + c) = *(const v8h*)(act2 + m * 512 + c);
                *(v8h*)(act2 + m * 512 + 256 + c) = *(const v8h*)(act3 + m * 256 + c);
            }
        }
        __syncthreads();

        // ---- Layer 1: a = [x|h1] @ W1^T, K=384, N=512; wave w -> z-tile w, cand-tile w+16 ----
        {
            v8f accZ = {}, accC = {};
            const _Float16* arow = act1 + m_a * 384;
            gf16_t* wz = wpt + W1_OFF + (size_t)(wv        * W1_NKT) * 512 + lane * 16;
            gf16_t* wc = wpt + W1_OFF + (size_t)((wv + 16) * W1_NKT) * 512 + lane * 16;
#pragma unroll 2
            for (int kt = 0; kt < 12; ++kt) {
                v16h a  = load_afrag(arow, kt, lane);
                v16h bz = *(gv16h_t*)(wz + kt * 512);
                v16h bc = *(gv16h_t*)(wc + kt * 512);
                accZ = wmma_f16(a, bz, accZ);
                accC = wmma_f16(a, bc, accC);
            }
#pragma unroll
            for (int r = 0; r < 8; ++r) {
                int m = mbase + r;
                float z  = 0.5f * (softsign(accZ[r] + biasZ1) + 1.0f);
                float cd = softsign(accC[r] + biasC1);
                float hold = (float)act1[m * 384 + 128 + n_t];
                act2[m * 512 + n_t] = (_Float16)(z * hold + (1.0f - z) * cd);
            }
        }
        __syncthreads();

        // ---- Layer 2: a = [h1_new|h2] @ W2^T, K=512, N=512 ----
        {
            v8f accZ = {}, accC = {};
            const _Float16* arow = act2 + m_a * 512;
            gf16_t* wz = wpt + W2_OFF + (size_t)(wv        * W2_NKT) * 512 + lane * 16;
            gf16_t* wc = wpt + W2_OFF + (size_t)((wv + 16) * W2_NKT) * 512 + lane * 16;
#pragma unroll 2
            for (int kt = 0; kt < 16; ++kt) {
                v16h a  = load_afrag(arow, kt, lane);
                v16h bz = *(gv16h_t*)(wz + kt * 512);
                v16h bc = *(gv16h_t*)(wc + kt * 512);
                accZ = wmma_f16(a, bz, accZ);
                accC = wmma_f16(a, bc, accC);
            }
#pragma unroll
            for (int r = 0; r < 8; ++r) {
                int m = mbase + r;
                float z  = 0.5f * (softsign(accZ[r] + biasZ2) + 1.0f);
                float cd = softsign(accC[r] + biasC2);
                float hold = (float)act2[m * 512 + 256 + n_t];
                act3[m * 256 + n_t] = (_Float16)(z * hold + (1.0f - z) * cd);
            }
        }
        __syncthreads();

        // ---- Layer 3: h3 = relu(h2_new @ W3^T + b3), K=256, N=128 ----
        if (wv < 8) {
            v8f acc = {};
            const _Float16* arow = act3 + m_a * 256;
            gf16_t* w3t = wpt + W3_OFF + (size_t)(wv * W3_NKT) * 512 + lane * 16;
#pragma unroll 2
            for (int kt = 0; kt < 8; ++kt) {
                v16h a = load_afrag(arow, kt, lane);
                v16h b = *(gv16h_t*)(w3t + kt * 512);
                acc = wmma_f16(a, b, acc);
            }
            int n = wv * 16 + ncol;
#pragma unroll
            for (int r = 0; r < 8; ++r) {
                float v = fmaxf(acc[r] + bias3, 0.0f);
                act4[(mbase + r) * 128 + n] = (_Float16)v;
            }
        }
        __syncthreads();

        // ---- Layer 4 + log_softmax: K=128, N=10 (padded to 16), wave 0 only ----
        if (wv == 0) {
            v8f acc = {};
            const _Float16* arow = act4 + m_a * 128;
            gf16_t* w4t = wpt + W4_OFF + lane * 16;
#pragma unroll 1
            for (int kt = 0; kt < 4; ++kt) {
                v16h a = load_afrag(arow, kt, lane);
                v16h b = *(gv16h_t*)(w4t + kt * 512);
                acc = wmma_f16(a, b, acc);
            }
#pragma unroll
            for (int r = 0; r < 8; ++r)
                smax[(mbase + r) * 16 + ncol] = acc[r] + bias4;
            // per-wave LDS ops are in-order; lanes 0..15 each reduce one row
            if (lane < 16) {
                const float* rowv = smax + lane * 16;
                float mx = rowv[0];
#pragma unroll
                for (int c = 1; c < NCLS; ++c) mx = fmaxf(mx, rowv[c]);
                float s = 0.0f;
#pragma unroll
                for (int c = 0; c < NCLS; ++c) s += __expf(rowv[c] - mx);
                float lse = mx + __logf(s);
                float* op = out + ((size_t)(bbase + lane) * TSTEPS + t) * NCLS;
#pragma unroll
                for (int c = 0; c < NCLS; ++c) op[c] = rowv[c] - lse;
            }
        }
        __syncthreads();
    }
}

extern "C" void kernel_launch(void* const* d_in, const int* in_sizes, int n_in,
                              void* d_out, int out_size, void* d_ws, size_t ws_size,
                              hipStream_t stream)
{
    (void)in_sizes; (void)n_in; (void)out_size; (void)ws_size;
    const float* x   = (const float*)d_in[0];
    const float* h1i = (const float*)d_in[1];
    const float* h2i = (const float*)d_in[2];
    const float* W1f = (const float*)d_in[3];
    const float* b1  = (const float*)d_in[4];
    const float* W2f = (const float*)d_in[5];
    const float* b2  = (const float*)d_in[6];
    const float* W3f = (const float*)d_in[7];
    const float* b3  = (const float*)d_in[8];
    const float* W4f = (const float*)d_in[9];
    const float* b4  = (const float*)d_in[10];
    float*    out = (float*)d_out;
    _Float16* wp  = (_Float16*)d_ws;   // needs WTOTAL*2 ~ 964 KB of workspace

    pack_weights<<<512, 256, 0, stream>>>(W1f, W2f, W3f, W4f, wp);
    egru_forward<<<16, 512, 0, stream>>>(x, h1i, h2i, b1, b2, b3, b4, wp, out);
}
